// ContactMapHead_83846351553194
// MI455X (gfx1250) — matrix operand order repaired and checked by reading, not compiled, not run on hardware
//
#include <hip/hip_runtime.h>
#include <hip/hip_bf16.h>

typedef __attribute__((ext_vector_type(16))) __bf16 v16bf;
typedef __attribute__((ext_vector_type(8)))  __bf16 v8bf;
typedef __attribute__((ext_vector_type(8)))  float  v8f;
typedef __attribute__((ext_vector_type(4)))  float  v4f;

#define BATCH   4
#define SEQ     2048
#define D_MODEL 1024
#define D_PROJ  256
#define ROWS    (BATCH * SEQ)          // 8192 rows through the projection

// ---------------------------------------------------------------------------
// Kernel 1: convert hidden_states (f32 [4,2048,1024]) and proj_w (f32
// [256,1024]) to bf16 in workspace. 8 elements per thread, vectorized.
// ---------------------------------------------------------------------------
__global__ __launch_bounds__(256)
void cvt_f32_to_bf16_kernel(const float* __restrict__ hidden,
                            const float* __restrict__ projw,
                            __bf16* __restrict__ hidden_bf,
                            __bf16* __restrict__ projw_bf) {
    const size_t NH = (size_t)ROWS * D_MODEL;      // 8,388,608
    const size_t NW = (size_t)D_PROJ * D_MODEL;    //   262,144
    size_t i = ((size_t)blockIdx.x * blockDim.x + threadIdx.x) * 8;
    const float* src;
    __bf16* dst;
    if (i < NH) {
        src = hidden + i;
        dst = hidden_bf + i;
    } else {
        size_t j = i - NH;
        if (j >= NW) return;
        src = projw + j;
        dst = projw_bf + j;
    }
    v4f f0 = *(const v4f*)(src);
    v4f f1 = *(const v4f*)(src + 4);
    v8bf o;
#pragma unroll
    for (int e = 0; e < 4; ++e) {
        o[e]     = (__bf16)f0[e];
        o[e + 4] = (__bf16)f1[e];
    }
    *(v8bf*)dst = o;
}

// ---------------------------------------------------------------------------
// Per-lane WMMA operand loaders (ISA 7.12.2 layouts, wave32).
//
// A (16x32 bf16, MxK): lanes 0-15 -> M=lane,    K in {0..7} U {16..23};
//                      lanes 16-31 -> M=lane-16, K in {8..15} U {24..31}.
// Row-major source: row (tile_m + lane&15), two contiguous 16B loads at
// k0 + 8*(lane>>4) and +16 elements (pointer pre-offset by half*8).
//
// B (32x16 bf16, KxN): lanes 0-15 -> N=lane, K in {0..7} U {16..23};
// lanes 16-31 -> N=lane-16, K in {8..15} U {24..31}. For B = H^T this is
// row (tile_n + lane&15) of H, two 16B loads (pointer pre-offset by half*8).
// ---------------------------------------------------------------------------
__device__ __forceinline__ v16bf load_a_tile(const __bf16* __restrict__ arow, int k0) {
    v8bf lo = *(const v8bf*)(arow + k0);
    v8bf hi = *(const v8bf*)(arow + k0 + 16);
    return __builtin_shufflevector(lo, hi, 0, 1, 2, 3, 4, 5, 6, 7,
                                   8, 9, 10, 11, 12, 13, 14, 15);
}

__device__ __forceinline__ v16bf load_b_tile(const __bf16* __restrict__ brow, int k0) {
    v8bf lo = *(const v8bf*)(brow + k0);
    v8bf hi = *(const v8bf*)(brow + k0 + 16);
    return __builtin_shufflevector(lo, hi, 0, 1, 2, 3, 4, 5, 6, 7,
                                   8, 9, 10, 11, 12, 13, 14, 15);
}

// ---------------------------------------------------------------------------
// Kernel 2: h = relu(X @ W^T + b), X bf16 [8192,1024], W bf16 [256,1024],
// h stored bf16 [8192,256]. One wave per 32x32 output block (2x2 WMMA tiles,
// operand fragments reused 2x). 512x16 tiles -> 256x8 blocks = 2048 waves.
// ---------------------------------------------------------------------------
__global__ __launch_bounds__(256)
void proj_relu_wmma_kernel(const __bf16* __restrict__ Xb,
                           const __bf16* __restrict__ Wb,
                           const float* __restrict__ bias,
                           __bf16* __restrict__ Hb) {
    const int lane = threadIdx.x & 31;
    const int wave = threadIdx.x >> 5;
    const int blk  = blockIdx.x * 8 + wave;   // 0..2047
    const int mtg  = blk >> 3;                // 0..255  (32-row group)
    const int ntg  = blk & 7;                 // 0..7    (32-col group)
    const int half = lane >> 4;
    const int l15  = lane & 15;

    const __bf16* arow[2];
    const __bf16* brow[2];
#pragma unroll
    for (int t = 0; t < 2; ++t) {
        arow[t] = Xb + (size_t)(mtg * 32 + t * 16 + l15) * D_MODEL + half * 8;
        brow[t] = Wb + (size_t)(ntg * 32 + t * 16 + l15) * D_MODEL + half * 8;
    }

    v8f acc[2][2] = {};
#pragma unroll 2
    for (int k0 = 0; k0 < D_MODEL; k0 += 32) {
        v16bf a0 = load_a_tile(arow[0], k0);
        v16bf a1 = load_a_tile(arow[1], k0);
        v16bf b0 = load_b_tile(brow[0], k0);
        v16bf b1 = load_b_tile(brow[1], k0);
        acc[0][0] = __builtin_amdgcn_wmma_f32_16x16x32_bf16(false, a0, false, b0, (short)0, acc[0][0], false, false);
        acc[0][1] = __builtin_amdgcn_wmma_f32_16x16x32_bf16(false, a0, false, b1, (short)0, acc[0][1], false, false);
        acc[1][0] = __builtin_amdgcn_wmma_f32_16x16x32_bf16(false, a1, false, b0, (short)0, acc[1][0], false, false);
        acc[1][1] = __builtin_amdgcn_wmma_f32_16x16x32_bf16(false, a1, false, b1, (short)0, acc[1][1], false, false);
    }

    // C/D layout: VGPR v holds (M = v + 8*half, N = lane&15).
#pragma unroll
    for (int tj = 0; tj < 2; ++tj) {
        const int ncol = ntg * 32 + tj * 16 + l15;
        const float bv = bias[ncol];
#pragma unroll
        for (int ti = 0; ti < 2; ++ti) {
#pragma unroll
            for (int v = 0; v < 8; ++v) {
                float x = acc[ti][tj][v] + bv;
                x = x > 0.0f ? x : 0.0f;
                const int m = mtg * 32 + ti * 16 + half * 8 + v;
                Hb[(size_t)m * D_PROJ + ncol] = (__bf16)x;
            }
        }
    }
}

// ---------------------------------------------------------------------------
// Kernel 3: scores[b] = (h[b] @ h[b]^T) * clf_w + clf_b, f32 [4,2048,2048].
// One wave per 64x64 output block: 4x4 WMMA tiles, 16 WMMAs per K-step with
// every loaded fragment reused 4x -> 4 KB of L2 reads per 1 KB of output
// (256 MB total vs 67 MB of stores). Output is streamed with non-temporal
// stores so the 4 MB h matrix stays resident in L2.
// Blocks: 4 * 32 * 32 = 4096 waves -> 512 blocks x 8 waves.
// ---------------------------------------------------------------------------
__global__ __launch_bounds__(256)
void scores_wmma_kernel(const __bf16* __restrict__ Hb,
                        const float* __restrict__ clf_w,
                        const float* __restrict__ clf_b,
                        float* __restrict__ out) {
    const int lane = threadIdx.x & 31;
    const int wave = threadIdx.x >> 5;
    const int blk  = blockIdx.x * 8 + wave;   // 0..4095
    const int b    = blk >> 10;               // 0..3
    const int r    = blk & 1023;
    const int it   = r >> 5;                  // 0..31 (64-row block)
    const int jt   = r & 31;                  // 0..31 (64-col block)
    const int half = lane >> 4;
    const int l15  = lane & 15;

    const __bf16* hb = Hb + (size_t)b * SEQ * D_PROJ;
    const __bf16* arow[4];
    const __bf16* brow[4];
#pragma unroll
    for (int t = 0; t < 4; ++t) {
        arow[t] = hb + (size_t)(it * 64 + t * 16 + l15) * D_PROJ + half * 8;
        brow[t] = hb + (size_t)(jt * 64 + t * 16 + l15) * D_PROJ + half * 8;
    }

    v8f acc[4][4] = {};
#pragma unroll 2
    for (int k0 = 0; k0 < D_PROJ; k0 += 32) {
        v16bf a[4], bm[4];
#pragma unroll
        for (int t = 0; t < 4; ++t) a[t]  = load_a_tile(arow[t], k0);
#pragma unroll
        for (int t = 0; t < 4; ++t) bm[t] = load_b_tile(brow[t], k0);
#pragma unroll
        for (int ti = 0; ti < 4; ++ti)
#pragma unroll
            for (int tj = 0; tj < 4; ++tj)
                acc[ti][tj] = __builtin_amdgcn_wmma_f32_16x16x32_bf16(
                    false, a[ti], false, bm[tj], (short)0, acc[ti][tj], false, false);
    }

    const float scale = clf_w[0];
    const float cb    = clf_b[0];
    // Base: row (it*64 + half*8), col (jt*64 + l15).
    float* obase = out + (size_t)b * SEQ * SEQ
                       + (size_t)(it * 64 + half * 8) * SEQ
                       + (size_t)(jt * 64 + l15);
#pragma unroll
    for (int ti = 0; ti < 4; ++ti) {
#pragma unroll
        for (int v = 0; v < 8; ++v) {
            float* orow = obase + (size_t)(ti * 16 + v) * SEQ;
#pragma unroll
            for (int tj = 0; tj < 4; ++tj) {
                __builtin_nontemporal_store(acc[ti][tj][v] * scale + cb,
                                            orow + tj * 16);
            }
        }
    }
}

// ---------------------------------------------------------------------------
// Host launcher. Workspace layout (d_ws):
//   [0,            16777216)  hidden_bf16  (8192 x 1024 bf16)
//   [16777216,     17301504)  proj_w_bf16  ( 256 x 1024 bf16)
//   [17301504,     21495808)  h_bf16       (8192 x  256 bf16)
// ---------------------------------------------------------------------------
extern "C" void kernel_launch(void* const* d_in, const int* in_sizes, int n_in,
                              void* d_out, int out_size, void* d_ws, size_t ws_size,
                              hipStream_t stream) {
    (void)in_sizes; (void)n_in; (void)out_size; (void)ws_size;

    const float* hidden = (const float*)d_in[0];
    const float* projw  = (const float*)d_in[1];
    const float* projb  = (const float*)d_in[2];
    const float* clfw   = (const float*)d_in[3];
    const float* clfb   = (const float*)d_in[4];
    float* out = (float*)d_out;

    char* ws = (char*)d_ws;
    __bf16* hidden_bf = (__bf16*)(ws);
    __bf16* projw_bf  = (__bf16*)(ws + 16777216);
    __bf16* h_bf      = (__bf16*)(ws + 16777216 + 524288);

    // 1) f32 -> bf16 conversions: (8388608 + 262144)/8 = 1,081,344 threads.
    {
        const int threads = 256;
        const int total   = (ROWS * D_MODEL + D_PROJ * D_MODEL) / 8;
        const int blocks  = (total + threads - 1) / threads;   // 4224
        cvt_f32_to_bf16_kernel<<<blocks, threads, 0, stream>>>(
            hidden, projw, hidden_bf, projw_bf);
    }

    // 2) Projection + ReLU: 2048 waves (32x32 blocks) -> 256 blocks.
    proj_relu_wmma_kernel<<<256, 256, 0, stream>>>(hidden_bf, projw_bf, projb, h_bf);

    // 3) Scores: 4096 waves (64x64 blocks) -> 512 blocks.
    scores_wmma_kernel<<<512, 256, 0, stream>>>(h_bf, clfw, clfb, out);
}
